// SATRot_4741643895473
// MI455X (gfx1250) — compile-verified, matrix operand order checked by reading, not run on hardware
//
#include <hip/hip_runtime.h>
#include <hip/hip_bf16.h>

typedef __attribute__((ext_vector_type(16))) _Float16 v16h;
typedef __attribute__((ext_vector_type(8)))  float    v8f;
typedef __attribute__((ext_vector_type(4)))  float    v4f;

#define B_   256
#define S_   11
#define D_   384
#define NH_  12
#define H_   128

// ---------------------------------------------------------------------------
// Fragment load: 16 f16 elements per lane for v_wmma_f32_16x16x32_f16.
// Per ISA layout (16-bit A 16x32): lane&15 = matrix row, lane>>4 selects the
// K-half (0 or 8). Elements 0..7 = K p+0..7, elements 8..15 = K p+16..23.
// Source is fp32; convert in registers (v_cvt_pk_f16_f32) to skip staging.
// ---------------------------------------------------------------------------
__device__ __forceinline__ v16h load_frag_f32(const float* __restrict__ p) {
  v4f x0 = *(const v4f*)(p);
  v4f x1 = *(const v4f*)(p + 4);
  v4f x2 = *(const v4f*)(p + 16);
  v4f x3 = *(const v4f*)(p + 20);
  v16h h;
#pragma unroll
  for (int i = 0; i < 4; ++i) {
    h[i]      = (_Float16)x0[i];
    h[i + 4]  = (_Float16)x1[i];
    h[i + 8]  = (_Float16)x2[i];
    h[i + 12] = (_Float16)x3[i];
  }
  return h;
}

__device__ __forceinline__ v8f wmma_step(v16h a, v16h b, v8f c) {
  return __builtin_amdgcn_wmma_f32_16x16x32_f16(false, a, false, b,
                                                (short)0, c, false, false);
}

// Straight-line (templated) epilogue for one 16x16 sub-tile.
template <bool RESID, bool RELU>
__device__ __forceinline__ void store_tile(float* __restrict__ C,
                                           const float* __restrict__ bias,
                                           const float* __restrict__ resid,
                                           v8f acc, int gm0, int gn, int N) {
  float bv = bias[gn];
#pragma unroll
  for (int j = 0; j < 8; ++j) {
    float v = acc[j] + bv;
    if (RESID) v += resid[(size_t)(gm0 + j) * N + gn];
    if (RELU)  v = fmaxf(v, 0.0f);
    C[(size_t)(gm0 + j) * N + gn] = v;
  }
}

// ---------------------------------------------------------------------------
// Generic GEMM: C[M,N] = A[M,K] @ W[N,K]^T + bias (+ resid) (+ relu)
// Register-blocked: one wave32 owns a 32x32 C tile = 4 WMMA accumulators.
// Per K-chunk of 32: 2 A-frags + 2 B-frags feed 4 back-to-back WMMAs
// (2x the FLOPs/byte of a 16x16 tile). K-chunk 8 ahead (k0+256) is prefetched into L2
// via global_prefetch_b8. Requires M%32==0, N%32==0, K%32==0 (all call
// sites satisfy this).
// ---------------------------------------------------------------------------
template <bool RESID, bool RELU>
__global__ __launch_bounds__(256)
void gemm_wmma_f16(const float* __restrict__ A, const float* __restrict__ W,
                   const float* __restrict__ bias, const float* __restrict__ resid,
                   float* __restrict__ C, int M, int N, int K) {
  int lane   = threadIdx.x & 31;
  int wave   = threadIdx.x >> 5;
  int tilesN = N >> 5;
  int tiles  = (M >> 5) * tilesN;
  int tile   = blockIdx.x * 8 + wave;     // 8 waves per block (wave32)
  if (tile >= tiles) return;              // wave-uniform exit: EXEC stays full
  int tm = tile / tilesN;
  int tn = tile - tm * tilesN;
  int row   = lane & 15;
  int khalf = (lane >> 4) << 3;           // 0 or 8
  const float* aP0 = A + (size_t)(tm * 32 + row) * K + khalf;
  const float* aP1 = aP0 + (size_t)16 * K;
  const float* wP0 = W + (size_t)(tn * 32 + row) * K + khalf;
  const float* wP1 = wP0 + (size_t)16 * K;
  v8f c00 = {}, c01 = {}, c10 = {}, c11 = {};
  for (int k0 = 0; k0 < K; k0 += 32) {
    __builtin_prefetch(aP0 + k0 + 256, 0, 1);   // global_prefetch_b8: next chunks
    __builtin_prefetch(wP0 + k0 + 256, 0, 1);
    v16h a0 = load_frag_f32(aP0 + k0);
    v16h a1 = load_frag_f32(aP1 + k0);
    v16h b0 = load_frag_f32(wP0 + k0);
    v16h b1 = load_frag_f32(wP1 + k0);
    c00 = wmma_step(a0, b0, c00);
    c01 = wmma_step(a0, b1, c01);
    c10 = wmma_step(a1, b0, c10);
    c11 = wmma_step(a1, b1, c11);
  }
  // C/D layout: lane&15 = N column, VGPR j = M row j (+8 for upper lanes)
  int mhalf = (lane >> 4) << 3;
  int gm0 = tm * 32 + mhalf;
  int gn0 = tn * 32 + (lane & 15);
  store_tile<RESID, RELU>(C, bias, resid, c00, gm0,      gn0,      N);
  store_tile<RESID, RELU>(C, bias, resid, c01, gm0,      gn0 + 16, N);
  store_tile<RESID, RELU>(C, bias, resid, c10, gm0 + 16, gn0,      N);
  store_tile<RESID, RELU>(C, bias, resid, c11, gm0 + 16, gn0 + 16, N);
}

// ---------------------------------------------------------------------------
// Conv branch: 7x7 stride-4 SAME conv (pad_lo=1) on the 128x128 window around
// pos0, ReLU, spatial mean -> (B,128). One block per image, thread = channel.
// ---------------------------------------------------------------------------
__global__ void conv_mean_k(const float* __restrict__ img, const int* __restrict__ pos0,
                            const float* __restrict__ cw, const float* __restrict__ cb,
                            float* __restrict__ outm) {
  int b  = blockIdx.x;
  int ch = threadIdx.x;  // 128 threads
  int y = min(max(pos0[0], 0), H_ - 1);
  int x = min(max(pos0[1], 0), H_ - 1);
  int y1 = max(0, y - 64), y2 = min(H_, y + 64);
  int x1 = max(0, x - 64), x2 = min(H_, x + 64);
  int ylen = y2 - y1, xlen = x2 - x1;
  const float* ib = img + (size_t)b * 4 * H_ * H_;  // 4-channel input, use first 3
  const float* wb = cw + (size_t)ch * 3 * 49;
  float bias = cb[ch];
  float sum = 0.f;
#pragma unroll 1
  for (int oy = 0; oy < 32; ++oy) {
#pragma unroll 1
    for (int ox = 0; ox < 32; ++ox) {
      float acc = bias;
#pragma unroll 1
      for (int kh = 0; kh < 7; ++kh) {
        int iy = oy * 4 - 1 + kh;
        if (iy < 0 || iy >= ylen) continue;
#pragma unroll 1
        for (int kw = 0; kw < 7; ++kw) {
          int ix = ox * 4 - 1 + kw;
          if (ix < 0 || ix >= xlen) continue;
          size_t pix = (size_t)(y1 + iy) * H_ + (x1 + ix);
          acc += ib[pix]         * wb[kh * 7 + kw];
          acc += ib[16384 + pix] * wb[49 + kh * 7 + kw];
          acc += ib[32768 + pix] * wb[98 + kh * 7 + kw];
        }
      }
      sum += fmaxf(acc, 0.f);
    }
  }
  outm[b * 128 + ch] = sum * (1.0f / 1024.0f);
}

// ---------------------------------------------------------------------------
// Gather the 10 small 24x24x3 windows per image into a dense (2560,1728) A
// matrix (flat order c,wy,wx) with the reference's edge-mask semantics.
// ---------------------------------------------------------------------------
__global__ void win1_gather_k(const float* __restrict__ img, const int* __restrict__ pos1,
                              float* __restrict__ out) {
  int idx = blockIdx.x * blockDim.x + threadIdx.x;
  if (idx >= 2560 * 1728) return;
  int k    = idx % 1728;
  int rowi = idx / 1728;
  int b = rowi / 10, j = rowi % 10;
  int y = min(max(pos1[2 * j], 0), H_ - 1);
  int x = min(max(pos1[2 * j + 1], 0), H_ - 1);
  int y1 = max(0, y - 12), y2 = min(H_, y + 12);
  int x1 = max(0, x - 12), x2 = min(H_, x + 12);
  int c = k / 576, rem = k - c * 576, wy = rem / 24, wx = rem - wy * 24;
  float v = 0.f;
  if (wy < (y2 - y1) && wx < (x2 - x1))
    v = img[((size_t)b * 4 + c) * 16384 + (size_t)(y1 + wy) * H_ + (x1 + wx)];
  out[idx] = v;
}

// ---------------------------------------------------------------------------
// x = feats + positional encoding (sin/cos over y-64, x-64, window-size).
// ---------------------------------------------------------------------------
__global__ void assemble_x_k(const float* __restrict__ f0, const float* __restrict__ f1,
                             const int* __restrict__ pos0, const int* __restrict__ pos1,
                             float* __restrict__ x) {
  int idx = blockIdx.x * blockDim.x + threadIdx.x;
  if (idx >= B_ * S_ * D_) return;
  int d  = idx % D_;
  int bt = idx / D_;
  int t  = bt % S_;
  int b  = bt / S_;
  int py, px; float wcoord;
  if (t == 0) { py = pos0[0]; px = pos0[1]; wcoord = 128.f; }
  else        { py = pos1[2 * (t - 1)]; px = pos1[2 * (t - 1) + 1]; wcoord = 24.f; }
  int seg = d >> 7;          // 0:y 1:x 2:window
  int r   = d & 127;
  int i2  = (r >> 1) << 1;
  float freq  = expf(-(float)i2 * (9.210340371976184f / 128.f));  // ln(1e4)/128
  float coord = (seg == 0) ? (float)(py - 64) : (seg == 1) ? (float)(px - 64) : wcoord;
  float ang   = coord * freq;
  float pe    = (r & 1) ? cosf(ang) : sinf(ang);
  float f = (t == 0) ? f0[b * D_ + d] : f1[(size_t)(b * 10 + t - 1) * D_ + d];
  x[idx] = f + pe;
}

// ---------------------------------------------------------------------------
// Tiny 11-token attention: one block per (b, head). qkv rows are 1152 wide
// with q/k/v at offsets 0/384/768 and head h at +h*32.
// ---------------------------------------------------------------------------
__global__ void attention_k(const float* __restrict__ qkv, float* __restrict__ o) {
  __shared__ float sc[121];
  int b = blockIdx.x / NH_;
  int h = blockIdx.x % NH_;
  int tid = threadIdx.x;  // 128 threads
  const float scale = 0.17677669529663687f;  // 1/sqrt(32)
  size_t base = (size_t)b * S_ * 1152 + h * 32;
  if (tid < 121) {
    int qi = tid / 11, ki = tid % 11;
    const float* qp = qkv + base + (size_t)qi * 1152;
    const float* kp = qkv + base + (size_t)ki * 1152 + 384;
    float acc = 0.f;
#pragma unroll
    for (int d = 0; d < 32; ++d) acc += qp[d] * kp[d];
    sc[tid] = acc * scale;
  }
  __syncthreads();
  if (tid < 11) {
    float mx = -1e30f;
    for (int k = 0; k < 11; ++k) mx = fmaxf(mx, sc[tid * 11 + k]);
    float e[11]; float s = 0.f;
    for (int k = 0; k < 11; ++k) { e[k] = expf(sc[tid * 11 + k] - mx); s += e[k]; }
    float inv = 1.f / s;
    for (int k = 0; k < 11; ++k) sc[tid * 11 + k] = e[k] * inv;
  }
  __syncthreads();
  if (tid < 352) {
    int qi = tid >> 5, d = tid & 31;
    float acc = 0.f;
    for (int ki = 0; ki < 11; ++ki)
      acc += sc[qi * 11 + ki] * qkv[base + (size_t)ki * 1152 + 768 + d];
    o[((size_t)b * S_ + qi) * D_ + h * 32 + d] = acc;
  }
}

// ---------------------------------------------------------------------------
// LayerNorm over D=384; one block (128 threads) per row. Residual was already
// folded into `in` by the GEMM epilogue.
// ---------------------------------------------------------------------------
__global__ void layernorm_k(const float* __restrict__ in, const float* __restrict__ g,
                            const float* __restrict__ bb, float* __restrict__ out) {
  __shared__ float rs[128], rs2[128];
  int row = blockIdx.x, tid = threadIdx.x;
  const float* p = in + (size_t)row * D_;
  float s = 0.f, s2 = 0.f;
  for (int d = tid; d < D_; d += 128) { float v = p[d]; s += v; s2 += v * v; }
  rs[tid] = s; rs2[tid] = s2;
  __syncthreads();
  for (int st = 64; st > 0; st >>= 1) {
    if (tid < st) { rs[tid] += rs[tid + st]; rs2[tid] += rs2[tid + st]; }
    __syncthreads();
  }
  float mean = rs[0] * (1.f / D_);
  float var  = rs2[0] * (1.f / D_) - mean * mean;
  float rstd = rsqrtf(var + 1e-5f);
  for (int d = tid; d < D_; d += 128)
    out[(size_t)row * D_ + d] = (p[d] - mean) * rstd * g[d] + bb[d];
}

__global__ void token_mean_k(const float* __restrict__ x, float* __restrict__ m) {
  int idx = blockIdx.x * blockDim.x + threadIdx.x;
  if (idx >= B_ * D_) return;
  int b = idx / D_, d = idx % D_;
  float s = 0.f;
  for (int t = 0; t < S_; ++t) s += x[((size_t)b * S_ + t) * D_ + d];
  m[idx] = s * (1.f / 11.f);
}

__device__ __forceinline__ void norm3(float& a, float& b, float& c) {
  float n = sqrtf(a * a + b * b + c * c);
  n = fmaxf(n, 1e-12f);
  a /= n; b /= n; c /= n;
}

// Final heads: tiny 384->6 / 384->2 dots + Gram-Schmidt rotation, per batch b.
__global__ void heads_k(const float* __restrict__ hr, const float* __restrict__ hu,
                        const float* __restrict__ rw2, const float* __restrict__ rb2,
                        const float* __restrict__ uw2, const float* __restrict__ ub2,
                        float* __restrict__ out) {
  int b = blockIdx.x * blockDim.x + threadIdx.x;
  if (b >= B_) return;
  const float* hrp = hr + (size_t)b * D_;
  const float* hup = hu + (size_t)b * D_;
  float r[6];
  for (int j = 0; j < 6; ++j) {
    float acc = rb2[j];
    for (int d = 0; d < D_; ++d) acc += hrp[d] * rw2[j * D_ + d];
    r[j] = acc;
  }
  float u[2];
  for (int j = 0; j < 2; ++j) {
    float acc = ub2[j];
    for (int d = 0; d < D_; ++d) acc += hup[d] * uw2[j * D_ + d];
    u[j] = 1.f / (1.f + expf(-acc));
  }
  float r1x = r[0], r1y = r[1], r1z = r[2];
  float r2x = r[3], r2y = r[4], r2z = r[5];
  float dp = r1x * r2x + r1y * r2y + r1z * r2z;
  r2x -= dp * r1x; r2y -= dp * r1y; r2z -= dp * r1z;
  float r3x = r1y * r2z - r1z * r2y;
  float r3y = r1z * r2x - r1x * r2z;
  float r3z = r1x * r2y - r1y * r2x;
  norm3(r1x, r1y, r1z); norm3(r2x, r2y, r2z); norm3(r3x, r3y, r3z);
  out[b * 2 + 0] = u[0];
  out[b * 2 + 1] = u[1];
  float* R = out + 2 * B_;
  R[b * 9 + 0] = r1x; R[b * 9 + 1] = r1y; R[b * 9 + 2] = r1z;
  R[b * 9 + 3] = r2x; R[b * 9 + 4] = r2y; R[b * 9 + 5] = r2z;
  R[b * 9 + 6] = r3x; R[b * 9 + 7] = r3y; R[b * 9 + 8] = r3z;
}

extern "C" void kernel_launch(void* const* d_in, const int* in_sizes, int n_in,
                              void* d_out, int out_size, void* d_ws, size_t ws_size,
                              hipStream_t stream) {
  const float* rgb    = (const float*)d_in[0];
  const int*   pos0   = (const int*)d_in[1];
  const int*   pos1   = (const int*)d_in[2];
  const float* conv_w = (const float*)d_in[3];
  const float* conv_b = (const float*)d_in[4];
  const float* ext_w  = (const float*)d_in[5];
  const float* ext_b  = (const float*)d_in[6];
  const float* proj_w = (const float*)d_in[7];
  const float* proj_b = (const float*)d_in[8];
  const float* Wqkv   = (const float*)d_in[9];
  const float* bqkv   = (const float*)d_in[10];
  const float* Wo     = (const float*)d_in[11];
  const float* bo     = (const float*)d_in[12];
  const float* ln1g   = (const float*)d_in[13];
  const float* ln1b   = (const float*)d_in[14];
  const float* ln2g   = (const float*)d_in[15];
  const float* ln2b   = (const float*)d_in[16];
  const float* Wff1   = (const float*)d_in[17];
  const float* bff1   = (const float*)d_in[18];
  const float* Wff2   = (const float*)d_in[19];
  const float* bff2   = (const float*)d_in[20];
  const float* rw1    = (const float*)d_in[21];
  const float* rb1    = (const float*)d_in[22];
  const float* rw2    = (const float*)d_in[23];
  const float* rb2    = (const float*)d_in[24];
  const float* uw1    = (const float*)d_in[25];
  const float* ub1    = (const float*)d_in[26];
  const float* uw2    = (const float*)d_in[27];
  const float* ub2    = (const float*)d_in[28];
  float* out = (float*)d_out;
  (void)in_sizes; (void)n_in; (void)out_size; (void)ws_size;

  char* base = (char*)d_ws;
  size_t off = 0;
  auto alloc = [&](size_t bytes) -> float* {
    char* p = base + off;
    off += (bytes + 255) & ~(size_t)255;
    return (float*)p;
  };
  // `big` is shared by win1 gather (2560*1728) and FFN hidden (2816*2048):
  // they are never live at the same time.
  float* big   = alloc((size_t)2816 * 2048 * 4);
  float* f1    = alloc((size_t)2560 * D_ * 4);
  float* cmean = alloc((size_t)B_ * 128 * 4);
  float* f0    = alloc((size_t)B_ * D_ * 4);
  float* x     = alloc((size_t)B_ * S_ * D_ * 4);
  float* xt    = alloc((size_t)B_ * S_ * D_ * 4);
  float* obuf  = alloc((size_t)B_ * S_ * D_ * 4);
  float* qkv   = alloc((size_t)B_ * S_ * 1152 * 4);
  float* mbuf  = alloc((size_t)B_ * D_ * 4);
  float* hr    = alloc((size_t)B_ * D_ * 4);
  float* hu    = alloc((size_t)B_ * D_ * 4);

  auto gemm = [&](const float* A, const float* W, const float* bias, const float* resid,
                  float* C, int M, int N, int K, int relu) {
    int tiles  = (M / 32) * (N / 32);
    int blocks = (tiles + 7) / 8;
    if (resid)
      gemm_wmma_f16<true, false><<<blocks, 256, 0, stream>>>(A, W, bias, resid, C, M, N, K);
    else if (relu)
      gemm_wmma_f16<false, true><<<blocks, 256, 0, stream>>>(A, W, bias, nullptr, C, M, N, K);
    else
      gemm_wmma_f16<false, false><<<blocks, 256, 0, stream>>>(A, W, bias, nullptr, C, M, N, K);
  };

  // Feature extraction
  conv_mean_k<<<B_, 128, 0, stream>>>(rgb, pos0, conv_w, conv_b, cmean);
  gemm(cmean, ext_w, ext_b, nullptr, f0, B_, D_, 128, 0);
  {
    int total = 2560 * 1728;
    win1_gather_k<<<(total + 255) / 256, 256, 0, stream>>>(rgb, pos1, big);
  }
  gemm(big, proj_w, proj_b, nullptr, f1, 2560, D_, 1728, 0);
  {
    int total = B_ * S_ * D_;
    assemble_x_k<<<(total + 255) / 256, 256, 0, stream>>>(f0, f1, pos0, pos1, x);
  }

  // Transformer encoder layers
  for (int l = 0; l < 4; ++l) {
    gemm(x, Wqkv + (size_t)l * 1152 * D_, bqkv + l * 1152, nullptr, qkv,
         B_ * S_, 1152, D_, 0);
    attention_k<<<B_ * NH_, 128, 0, stream>>>(qkv, obuf);
    gemm(obuf, Wo + (size_t)l * D_ * D_, bo + l * D_, x, xt, B_ * S_, D_, D_, 0);
    layernorm_k<<<B_ * S_, 128, 0, stream>>>(xt, ln1g + l * D_, ln1b + l * D_, x);
    gemm(x, Wff1 + (size_t)l * 2048 * D_, bff1 + l * 2048, nullptr, big,
         B_ * S_, 2048, D_, 1);
    gemm(big, Wff2 + (size_t)l * D_ * 2048, bff2 + l * D_, x, xt, B_ * S_, D_, 2048, 0);
    layernorm_k<<<B_ * S_, 128, 0, stream>>>(xt, ln2g + l * D_, ln2b + l * D_, x);
  }

  // Heads
  token_mean_k<<<(B_ * D_ + 255) / 256, 256, 0, stream>>>(x, mbuf);
  gemm(mbuf, rw1, rb1, nullptr, hr, B_, D_, D_, 1);
  gemm(mbuf, uw1, ub1, nullptr, hu, B_, D_, D_, 1);
  heads_k<<<1, 256, 0, stream>>>(hr, hu, rw2, rb2, uw2, ub2, out);
}
